// RoomActor_88673894793688
// MI455X (gfx1250) — matrix operand order repaired and checked by reading, not compiled, not run on hardware
//
#include <hip/hip_runtime.h>
#include <hip/hip_bf16.h>

#define HID   128
#define EMB   64
#define COND  192
#define HTOT  320   // HID + COND
#define K1TOT 640   // 2*HTOT
#define EPB   32    // edges per block in the WMMA edge-conv kernel
#define SA_STRIDE 648   // 640 + 8 halves pad -> 4-dword bank skew
#define ST_STRIDE 136   // 128 + 8 halves pad
#define NPB   16    // nodes per block for node-level MLP kernels
#define ENC_NEG_INF 0x007FFFFFu   // enc(-inf)

typedef _Float16 v16h __attribute__((ext_vector_type(16)));
typedef _Float16 v8h  __attribute__((ext_vector_type(8)));
typedef float    v8f  __attribute__((ext_vector_type(8)));

union V16H { v16h v; v8h h[2]; };

// Order-preserving float<->uint mapping so we can use deterministic u32 atomic max.
__device__ __forceinline__ unsigned enc_f32(float x) {
  unsigned b = __float_as_uint(x);
  return (b & 0x80000000u) ? ~b : (b | 0x80000000u);
}
__device__ __forceinline__ float dec_f32(unsigned y) {
  unsigned b = (y & 0x80000000u) ? (y & 0x7FFFFFFFu) : ~y;
  return __uint_as_float(b);
}

// ---------------------------------------------------------------- utilities
__global__ void ra_fill_u32_kernel(unsigned* __restrict__ p, unsigned v, unsigned n) {
  unsigned i = blockIdx.x * blockDim.x + threadIdx.x;
  if (i < n) p[i] = v;
}

__global__ void ra_f32_to_f16_kernel(const float* __restrict__ s, _Float16* __restrict__ d, int n) {
  int i = blockIdx.x * blockDim.x + threadIdx.x;
  if (i < n) d[i] = (_Float16)s[i];
}

// ---------------------------------------------------------------- tiny precompute:
// class_table[c] = tanh(tanh(emb_table[c]) @ emb_w + emb_b)       (CLS x 64)
// wall_feat[b]   = tanh(mlp2(wall[b]))                            (B x 64)
__global__ __launch_bounds__(64) void ra_prep_kernel(
    const float* __restrict__ emb_table, const float* __restrict__ emb_w,
    const float* __restrict__ emb_b,
    const float* __restrict__ wall, const float* __restrict__ wall_w1,
    const float* __restrict__ wall_b1, const float* __restrict__ wall_w2,
    const float* __restrict__ wall_b2,
    float* __restrict__ class_table, float* __restrict__ wall_feat, int CLS, int B) {
  __shared__ float s[EMB];
  const int j = threadIdx.x;
  const int blk = blockIdx.x;
  if (blk < CLS) {
    s[j] = tanhf(emb_table[blk * EMB + j]);
    __syncthreads();
    float acc = emb_b[j];
    for (int k = 0; k < EMB; ++k) acc += s[k] * emb_w[k * EMB + j];
    class_table[blk * EMB + j] = tanhf(acc);
  } else {
    const int b = blk - CLS;
    const float w = wall[b];
    s[j] = tanhf(w * wall_w1[j] + wall_b1[j]);
    __syncthreads();
    float acc = wall_b2[j];
    for (int k = 0; k < EMB; ++k) acc += s[k] * wall_w2[k * EMB + j];
    wall_feat[b * EMB + j] = tanhf(acc);
  }
}

// ---------------------------------------------------------------- node init:
// h[n] = [ tanh(mlp2([x,tar])) (128) | class (64) | wall (64) | tanh(mlp2(geo)) (64) ]  (f16)
__global__ __launch_bounds__(128) void ra_node_init_kernel(
    const float* __restrict__ x, const float* __restrict__ tar,
    const int* __restrict__ category, const float* __restrict__ geo,
    const int* __restrict__ batch_idx,
    const float* __restrict__ iw1, const float* __restrict__ ib1,
    const float* __restrict__ iw2, const float* __restrict__ ib2,
    const float* __restrict__ gw1, const float* __restrict__ gb1,
    const float* __restrict__ gw2, const float* __restrict__ gb2,
    const float* __restrict__ class_table, const float* __restrict__ wall_feat,
    _Float16* __restrict__ h, int N) {
  __shared__ float sin7[8];
  __shared__ float s1[HID];
  __shared__ float g1[EMB];
  const int tid = threadIdx.x;
  for (int it = 0; it < NPB; ++it) {
    const int n = blockIdx.x * NPB + it;   // uniform across block
    if (n >= N) break;
    __syncthreads();
    if (tid < 4) sin7[tid] = x[(size_t)n * 4 + tid];
    else if (tid < 7) sin7[tid] = tar[(size_t)n * 3 + (tid - 4)];
    if (tid < EMB) {
      const float gx = geo[(size_t)n * 2 + 0], gy = geo[(size_t)n * 2 + 1];
      g1[tid] = tanhf(gx * gw1[tid] + gy * gw1[EMB + tid] + gb1[tid]);
    }
    __syncthreads();
    float a = ib1[tid];
#pragma unroll
    for (int k = 0; k < 7; ++k) a += sin7[k] * iw1[k * HID + tid];
    s1[tid] = tanhf(a);
    __syncthreads();
    float o = ib2[tid];
    for (int k = 0; k < HID; ++k) o += s1[k] * iw2[k * HID + tid];
    _Float16* hr = h + (size_t)n * HTOT;
    hr[tid] = (_Float16)tanhf(o);
    if (tid < EMB) {
      float go = gb2[tid];
      for (int k = 0; k < EMB; ++k) go += g1[k] * gw2[k * EMB + tid];
      hr[HID + tid]           = (_Float16)class_table[category[n] * EMB + tid];
      hr[HID + EMB + tid]     = (_Float16)wall_feat[batch_idx[n] * EMB + tid];
      hr[HID + 2 * EMB + tid] = (_Float16)tanhf(go);
    }
  }
}

// ---------------------------------------------------------------- WMMA edge conv
// A-fragment for v_wmma_f32_16x16x32_f16 from an LDS row:
// lane<16: K = kb+0..7 and kb+16..23 ; lane>=16: K = kb+8..15 and kb+24..31
__device__ __forceinline__ v16h ra_load_a_lds(const _Float16* rowp, int off0) {
  V16H u;
  u.h[0] = *(const v8h*)(rowp + off0);
  u.h[1] = *(const v8h*)(rowp + 16 + off0);
  return u.v;
}

__global__ __launch_bounds__(256) void ra_edge_conv_kernel(
    const _Float16* __restrict__ h,
    const int* __restrict__ src, const int* __restrict__ dst,
    const _Float16* __restrict__ W1h, const float* __restrict__ b1,
    const _Float16* __restrict__ W2h, const float* __restrict__ b2,
    unsigned* __restrict__ out_enc, int E) {
  __shared__ __align__(16) _Float16 sA[EPB * SA_STRIDE];  // 32 x 640 (padded)
  __shared__ __align__(16) _Float16 sT[EPB * ST_STRIDE];  // 32 x 128 (padded)
  __shared__ int sDst[EPB];

  const int tid  = threadIdx.x;
  const int lane = tid & 31;
  const int wave = tid >> 5;
  const int e0   = blockIdx.x * EPB;

  // ---- stage A = [h[dst] | h[src]-h[dst]] into LDS (8 threads per edge)
  {
    const int r = tid >> 3;
    const int t = tid & 7;
    const int e = e0 + r;
    const bool valid = (e < E);
    const int es = valid ? src[e] : 0;
    const int ed = valid ? dst[e] : 0;
    if (t == 0) sDst[r] = valid ? ed : -1;
    const v8h* hs = (const v8h*)(h + (size_t)es * HTOT);
    const v8h* hd = (const v8h*)(h + (size_t)ed * HTOT);
#pragma unroll
    for (int c = 0; c < 5; ++c) {
      const int cc = t + c * 8;           // 40 chunks of 8 halves = 320
      const v8h hi = hd[cc];
      const v8h hj = hs[cc];
      *(v8h*)(sA + r * SA_STRIDE + cc * 8)        = hi;
      *(v8h*)(sA + r * SA_STRIDE + HTOT + cc * 8) = hj - hi;
    }
  }
  __syncthreads();

  const int n0   = wave * 16;             // output-column tile of this wave
  const int mrow = lane & 15;
  const int off0 = (lane < 16) ? 0 : 8;
  const float bias1 = b1[n0 + mrow];
  const float bias2 = b2[n0 + mrow];

  // ---- layer 1: (32 x 640) @ (640 x 128), each wave does 2 M-subtiles x its 16 cols
  v8f acc0 = {}; v8f acc1 = {};
  for (int kb = 0; kb < K1TOT; kb += 32) {
    if (kb + 32 < K1TOT)
      __builtin_prefetch(W1h + (size_t)(kb + 32 + lane) * HID + n0, 0, 1);
    v16h bf; // lane%16 = N col, elements = K in table order
#pragma unroll
    for (int s = 0; s < 16; ++s) bf[s] = W1h[(size_t)(kb + off0 + (s < 8 ? s : s + 8)) * HID + n0 + mrow];
    const v16h a0 = ra_load_a_lds(sA + mrow * SA_STRIDE + kb, off0);
    const v16h a1 = ra_load_a_lds(sA + (16 + mrow) * SA_STRIDE + kb, off0);
    acc0 = __builtin_amdgcn_wmma_f32_16x16x32_f16(false, a0, false, bf, (short)0, acc0, false, false);
    acc1 = __builtin_amdgcn_wmma_f32_16x16x32_f16(false, a1, false, bf, (short)0, acc1, false, false);
  }

  // bias + tanh -> f16 staged for layer 2 (D layout: lane%16 = N, vgpr r + 8*(lane>=16) = M)
#pragma unroll
  for (int r = 0; r < 8; ++r) {
    const int mm = r + off0;
    sT[mm        * ST_STRIDE + n0 + mrow] = (_Float16)tanhf(acc0[r] + bias1);
    sT[(16 + mm) * ST_STRIDE + n0 + mrow] = (_Float16)tanhf(acc1[r] + bias1);
  }
  __syncthreads();

  // ---- layer 2: (32 x 128) @ (128 x 128)
  v8f c0v = {}; v8f c1v = {};
#pragma unroll
  for (int kb = 0; kb < HID; kb += 32) {
    v16h bf;
#pragma unroll
    for (int s = 0; s < 16; ++s) bf[s] = W2h[(size_t)(kb + off0 + (s < 8 ? s : s + 8)) * HID + n0 + mrow];
    const v16h a0 = ra_load_a_lds(sT + mrow * ST_STRIDE + kb, off0);
    const v16h a1 = ra_load_a_lds(sT + (16 + mrow) * ST_STRIDE + kb, off0);
    c0v = __builtin_amdgcn_wmma_f32_16x16x32_f16(false, a0, false, bf, (short)0, c0v, false, false);
    c1v = __builtin_amdgcn_wmma_f32_16x16x32_f16(false, a1, false, bf, (short)0, c1v, false, false);
  }

  // ---- segment_max over dst via order-preserving u32 atomic max (deterministic)
#pragma unroll
  for (int r = 0; r < 8; ++r) {
    const int mm = r + off0;
    const int d0 = sDst[mm];
    if (d0 >= 0)
      atomicMax(out_enc + (size_t)d0 * HID + n0 + mrow, enc_f32(c0v[r] + bias2));
    const int d1 = sDst[16 + mm];
    if (d1 >= 0)
      atomicMax(out_enc + (size_t)d1 * HID + n0 + mrow, enc_f32(c1v[r] + bias2));
  }
}

// ---------------------------------------------------------------- h[:, :128] = tanh(fix(conv))
__global__ void ra_update_h_kernel(const unsigned* __restrict__ conv_enc,
                                   _Float16* __restrict__ h, int N) {
  const int i = blockIdx.x * blockDim.x + threadIdx.x;
  if (i >= N * HID) return;
  const int n = i / HID, j = i % HID;
  float v = dec_f32(conv_enc[i]);
  if (!__builtin_isfinite(v)) v = 0.0f;   // nodes with no incoming edges
  h[(size_t)n * HTOT + j] = (_Float16)tanhf(v);
}

// ---------------------------------------------------------------- tail MLP + mu/std
__global__ __launch_bounds__(128) void ra_tail_kernel(
    const _Float16* __restrict__ h, const unsigned* __restrict__ conv_enc,
    const float* __restrict__ tw1, const float* __restrict__ tb1,
    const float* __restrict__ tw2, const float* __restrict__ tb2,
    float* __restrict__ out, int N) {
  __shared__ float sh[HTOT];
  __shared__ float st1[HID];
  const int tid = threadIdx.x;
  for (int it = 0; it < NPB; ++it) {
    const int n = blockIdx.x * NPB + it;   // uniform
    if (n >= N) break;
    __syncthreads();
    for (int k = tid; k < HTOT; k += 128) {
      float v;
      if (k < HID) {
        v = dec_f32(conv_enc[(size_t)n * HID + k]);
        if (!__builtin_isfinite(v)) v = 0.0f;
        v = tanhf(v);
      } else {
        v = (float)h[(size_t)n * HTOT + k];   // cond part
      }
      sh[k] = v;
    }
    __syncthreads();
    float a = tb1[tid];
    for (int k = 0; k < HTOT; ++k) a += sh[k] * tw1[k * HID + tid];
    st1[tid] = tanhf(a);
    __syncthreads();
    if (tid < 6) {
      float o = tb2[tid];
      for (int k = 0; k < HID; ++k) o += st1[k] * tw2[k * 6 + tid];
      if (tid < 3) {
        out[(size_t)n * 3 + tid] = tanhf(o);                 // MAX_ACTION = 1.0
      } else {
        float ls = tanhf(o);
        ls = -5.0f + 0.5f * 7.0f * (ls + 1.0f);              // [-5, 2]
        out[(size_t)N * 3 + (size_t)n * 3 + (tid - 3)] = expf(ls);
      }
    }
  }
}

// ---------------------------------------------------------------- launch
extern "C" void kernel_launch(void* const* d_in, const int* in_sizes, int n_in,
                              void* d_out, int out_size, void* d_ws, size_t ws_size,
                              hipStream_t stream) {
  (void)n_in; (void)out_size; (void)ws_size;
  const float* x          = (const float*)d_in[0];
  const float* tar        = (const float*)d_in[1];
  const int*   category   = (const int*)d_in[2];
  const float* geo        = (const float*)d_in[3];
  const float* wall       = (const float*)d_in[4];
  const int*   batch_idx  = (const int*)d_in[5];
  const int*   edge_index = (const int*)d_in[6];
  const float* init_w1 = (const float*)d_in[7];
  const float* init_b1 = (const float*)d_in[8];
  const float* init_w2 = (const float*)d_in[9];
  const float* init_b2 = (const float*)d_in[10];
  const float* wall_w1 = (const float*)d_in[11];
  const float* wall_b1 = (const float*)d_in[12];
  const float* wall_w2 = (const float*)d_in[13];
  const float* wall_b2 = (const float*)d_in[14];
  const float* emb_table = (const float*)d_in[15];
  const float* emb_w   = (const float*)d_in[16];
  const float* emb_b   = (const float*)d_in[17];
  const float* geo_w1  = (const float*)d_in[18];
  const float* geo_b1  = (const float*)d_in[19];
  const float* geo_w2  = (const float*)d_in[20];
  const float* geo_b2  = (const float*)d_in[21];
  const float* m1_w1   = (const float*)d_in[22];
  const float* m1_b1   = (const float*)d_in[23];
  const float* m1_w2   = (const float*)d_in[24];
  const float* m1_b2   = (const float*)d_in[25];
  const float* m2_w1   = (const float*)d_in[26];
  const float* m2_b1   = (const float*)d_in[27];
  const float* m2_w2   = (const float*)d_in[28];
  const float* m2_b2   = (const float*)d_in[29];
  const float* tail_w1 = (const float*)d_in[30];
  const float* tail_b1 = (const float*)d_in[31];
  const float* tail_w2 = (const float*)d_in[32];
  const float* tail_b2 = (const float*)d_in[33];

  const int N   = in_sizes[2];
  const int Bw  = in_sizes[4];
  const int E   = in_sizes[6] / 2;
  const int CLS = in_sizes[15] / EMB;
  const int* srcp = edge_index;
  const int* dstp = edge_index + E;

  size_t off = 0;
  auto wsalloc = [&](size_t bytes) -> void* {
    void* p = (char*)d_ws + off;
    off += (bytes + 255) & ~(size_t)255;
    return p;
  };
  _Float16* h        = (_Float16*)wsalloc((size_t)N * HTOT * sizeof(_Float16));
  unsigned* conv     = (unsigned*)wsalloc((size_t)N * HID * sizeof(unsigned));
  float* class_table = (float*)wsalloc((size_t)CLS * EMB * sizeof(float));
  float* wall_feat   = (float*)wsalloc((size_t)Bw * EMB * sizeof(float));
  _Float16* w1h_m1   = (_Float16*)wsalloc((size_t)K1TOT * HID * sizeof(_Float16));
  _Float16* w2h_m1   = (_Float16*)wsalloc((size_t)HID * HID * sizeof(_Float16));
  _Float16* w1h_m2   = (_Float16*)wsalloc((size_t)K1TOT * HID * sizeof(_Float16));
  _Float16* w2h_m2   = (_Float16*)wsalloc((size_t)HID * HID * sizeof(_Float16));

  ra_prep_kernel<<<CLS + Bw, 64, 0, stream>>>(
      emb_table, emb_w, emb_b, wall, wall_w1, wall_b1, wall_w2, wall_b2,
      class_table, wall_feat, CLS, Bw);

  const int n1 = K1TOT * HID, n2 = HID * HID;
  ra_f32_to_f16_kernel<<<(n1 + 255) / 256, 256, 0, stream>>>(m1_w1, w1h_m1, n1);
  ra_f32_to_f16_kernel<<<(n2 + 255) / 256, 256, 0, stream>>>(m1_w2, w2h_m1, n2);
  ra_f32_to_f16_kernel<<<(n1 + 255) / 256, 256, 0, stream>>>(m2_w1, w1h_m2, n1);
  ra_f32_to_f16_kernel<<<(n2 + 255) / 256, 256, 0, stream>>>(m2_w2, w2h_m2, n2);

  ra_node_init_kernel<<<(N + NPB - 1) / NPB, 128, 0, stream>>>(
      x, tar, category, geo, batch_idx,
      init_w1, init_b1, init_w2, init_b2,
      geo_w1, geo_b1, geo_w2, geo_b2,
      class_table, wall_feat, h, N);

  const unsigned nConv = (unsigned)(N * HID);
  const int convBlocks = (E + EPB - 1) / EPB;

  ra_fill_u32_kernel<<<(nConv + 255) / 256, 256, 0, stream>>>(conv, ENC_NEG_INF, nConv);
  ra_edge_conv_kernel<<<convBlocks, 256, 0, stream>>>(h, srcp, dstp, w1h_m1, m1_b1, w2h_m1, m1_b2, conv, E);
  ra_update_h_kernel<<<(nConv + 255) / 256, 256, 0, stream>>>(conv, h, N);

  ra_fill_u32_kernel<<<(nConv + 255) / 256, 256, 0, stream>>>(conv, ENC_NEG_INF, nConv);
  ra_edge_conv_kernel<<<convBlocks, 256, 0, stream>>>(h, srcp, dstp, w1h_m2, m2_b1, w2h_m2, m2_b2, conv, E);

  ra_tail_kernel<<<(N + NPB - 1) / NPB, 128, 0, stream>>>(
      h, conv, tail_w1, tail_b1, tail_w2, tail_b2, (float*)d_out, N);
}
